// GBST_20779051778155
// MI455X (gfx1250) — matrix operand order repaired and checked by reading, not compiled, not run on hardware
//
#include <hip/hip_runtime.h>
#include <hip/hip_bf16.h>

#define B_  16
#define T_  2048
#define E_  512
#define K_  4
#define NEGINF (-3.402823466e38f)

// GEMM tiling
#define BM 128
#define BN 128
#define CH 32                   // K-elements (e) per chunk
#define NCH (E_ / CH)           // 16 chunks
#define LROW 80                 // padded LDS row stride (64B data + 16B pad, 16B aligned)
#define ASTG (BM * LROW)        // 10240 B per A stage buffer
#define BSTG (BN * LROW)        // 10240 B per B stage buffer
#define NSTAGE 3

typedef __attribute__((ext_vector_type(16))) __bf16 v16bf;
typedef __attribute__((ext_vector_type(8)))  float  v8f;

union Frag32 { uint4 q[2]; v16bf v; };

__device__ __forceinline__ unsigned short f2bf(float f) {
  unsigned int u = __float_as_uint(f);
  unsigned int r = (u + 0x7FFFu + ((u >> 16) & 1u)) >> 16;
  return (unsigned short)r;
}

// CDNA5 async global->LDS copy (16B per lane), tracked by ASYNCcnt.
__device__ __forceinline__ void async_copy16(unsigned lds_off, const void* g) {
  asm volatile("global_load_async_to_lds_b128 %0, %1, off"
               :: "v"(lds_off), "v"(g) : "memory");
}
__device__ __forceinline__ void wait_async0() {
  asm volatile("s_wait_asynccnt 0" ::: "memory");
}
__device__ __forceinline__ void wait_async4() {  // allow newest stage (4 copies) in flight
  asm volatile("s_wait_asynccnt 4" ::: "memory");
}

// ---------------- Kernel 0: pw_kernel (E,E) f32 -> bf16, row-major (o,e)
__global__ void __launch_bounds__(256) k_pw_bf16(const float* __restrict__ pw,
                                                 unsigned short* __restrict__ pwb) {
  int i = blockIdx.x * blockDim.x + threadIdx.x;   // over E*E/4 float4s
  float4 f = ((const float4*)pw)[i];
  ushort4 o;
  o.x = f2bf(f.x); o.y = f2bf(f.y); o.z = f2bf(f.z); o.w = f2bf(f.w);
  ((ushort4*)pwb)[i] = o;
}

// ---------------- Kernel 1: embedding gather + depthwise conv -> bf16 A[b*T+t, e]
__global__ void __launch_bounds__(128) k_conv(const int* __restrict__ seq,
                                              const float* __restrict__ emb,
                                              const float* __restrict__ dw,
                                              unsigned short* __restrict__ A) {
  const int bt = blockIdx.x;
  const int b = bt / T_, t = bt % T_;
  const int e0 = threadIdx.x * 4;            // 128 threads cover E=512
  float4 dwv[4];
  #pragma unroll
  for (int j = 0; j < 4; ++j) dwv[j] = ((const float4*)dw)[e0 + j];
  float4 acc = {0.f, 0.f, 0.f, 0.f};
  #pragma unroll
  for (int k = 0; k < K_; ++k) {
    int tk = t + k;
    if (tk < T_) {
      int s = seq[b * T_ + tk];
      float4 ev = *(const float4*)(emb + (size_t)s * E_ + e0);
      acc.x += ((const float*)&dwv[0])[k] * ev.x;
      acc.y += ((const float*)&dwv[1])[k] * ev.y;
      acc.z += ((const float*)&dwv[2])[k] * ev.z;
      acc.w += ((const float*)&dwv[3])[k] * ev.w;
    }
  }
  ushort4 o4;
  o4.x = f2bf(acc.x); o4.y = f2bf(acc.y); o4.z = f2bf(acc.z); o4.w = f2bf(acc.w);
  *(ushort4*)(A + (size_t)bt * E_ + e0) = o4;
}

// ---------------- Kernel 2: WMMA GEMM  C[m,o] = sum_e A[m,e] * pw[o,e]
// M=B*T=32768, N=E=512, K=E=512, bf16 in / f32 accumulate.
// 256 threads = 8 waves; block tile 128x128; wave tile 64(M)x32(N) = 8 WMMAs/chunk.
// 3-stage circular LDS pipeline fed by async global->LDS copies (ASYNCcnt).
__global__ void __launch_bounds__(256) k_gemm(const unsigned short* __restrict__ A,
                                              const unsigned short* __restrict__ Bm,
                                              float* __restrict__ C) {
  __shared__ __align__(16) char smem[NSTAGE * (ASTG + BSTG)];
  const unsigned ldsBase = (unsigned)(uintptr_t)&smem[0];  // flat->LDS: low 32 bits
  const unsigned ldsA = ldsBase;
  const unsigned ldsB = ldsBase + NSTAGE * ASTG;

  const int tid  = threadIdx.x;
  const int lane = tid & 31;
  const int wave = tid >> 5;
  const int wm   = wave & 1;              // 2 waves along M (64 rows each)
  const int wn   = wave >> 1;             // 4 waves along N (32 cols each)
  const int m_blk = blockIdx.x * BM;
  const int n_blk = blockIdx.y * BN;
  const int sel  = lane >> 4;             // 0: lanes 0-15, 1: lanes 16-31
  const int lr   = lane & 15;

  const char* Ag = (const char*)A  + (size_t)m_blk * (E_ * 2);
  const char* Bg = (const char*)Bm + (size_t)n_blk * (E_ * 2);

  // Staging: A = 512 16B segments, B = 512; each thread moves 2 A + 2 B segments.
  const int r0 = tid >> 2,          p0 = (tid & 3) * 16;       // rows/cols 0..63
  const int r1 = (tid + 256) >> 2,  p1 = p0;                   // rows/cols 64..127

  auto stage = [&](int kc, int buf) {
    const int kb = kc * 2;  // byte offset along e
    async_copy16(ldsA + buf * ASTG + r0 * LROW + p0, Ag + (size_t)r0 * (E_*2) + kb + p0);
    async_copy16(ldsA + buf * ASTG + r1 * LROW + p1, Ag + (size_t)r1 * (E_*2) + kb + p1);
    async_copy16(ldsB + buf * BSTG + r0 * LROW + p0, Bg + (size_t)r0 * (E_*2) + kb + p0);
    async_copy16(ldsB + buf * BSTG + r1 * LROW + p1, Bg + (size_t)r1 * (E_*2) + kb + p1);
  };

  v8f c[4][2] = {{{}, {}}, {{}, {}}, {{}, {}}, {{}, {}}};

  stage(0, 0);
  stage(CH, 1);

  int buf = 0;
  for (int kc = 0; kc < E_; kc += CH) {
    // Retire the stage we are about to consume; keep the newest one in flight.
    if (kc + CH < E_) wait_async4(); else wait_async0();
    __syncthreads();
    if (kc + 2 * CH < E_) {
      int nb = buf + 2; if (nb >= NSTAGE) nb -= NSTAGE;
      stage(kc + 2 * CH, nb);
    }

    // Preload ALL fragments for this chunk, then issue 8 back-to-back WMMAs.
    Frag32 bfr[2], afr[4];
    #pragma unroll
    for (int ni = 0; ni < 2; ++ni) {
      const char* bp = smem + NSTAGE * ASTG + buf * BSTG
                     + (wn * 32 + ni * 16 + lr) * LROW + sel * 32;
      bfr[ni].q[0] = *(const uint4*)(bp);
      bfr[ni].q[1] = *(const uint4*)(bp + 16);
    }
    #pragma unroll
    for (int mi = 0; mi < 4; ++mi) {
      const char* ap = smem + buf * ASTG
                     + (wm * 64 + mi * 16 + lr) * LROW + sel * 32;
      afr[mi].q[0] = *(const uint4*)(ap);
      afr[mi].q[1] = *(const uint4*)(ap + 16);
    }
    #pragma unroll
    for (int mi = 0; mi < 4; ++mi) {
      #pragma unroll
      for (int ni = 0; ni < 2; ++ni) {
        c[mi][ni] = __builtin_amdgcn_wmma_f32_16x16x32_bf16(
            false, afr[mi].v, false, bfr[ni].v, (short)0, c[mi][ni], false, false);
      }
    }

    ++buf; if (buf >= NSTAGE) buf -= NSTAGE;
  }

  // C/D layout: VGPR j -> M = j + sel*8, N = lane&15
  #pragma unroll
  for (int mi = 0; mi < 4; ++mi) {
    #pragma unroll
    for (int ni = 0; ni < 2; ++ni) {
      const int o = n_blk + wn * 32 + ni * 16 + lr;
      float* outp = C + (size_t)(m_blk + wm * 64 + mi * 16 + sel * 8) * E_ + o;
      #pragma unroll
      for (int j = 0; j < 8; ++j) outp[(size_t)j * E_] = c[mi][ni][j];
    }
  }
}

// ---------------- Kernel 3: segment means + masked softmax mix -> out[b,t,e]
__global__ void __launch_bounds__(128) k_combine(const int* __restrict__ seq,
                                                 const int* __restrict__ gidp,
                                                 const float* __restrict__ xs,
                                                 const float* __restrict__ sw,
                                                 const float* __restrict__ sb,
                                                 float* __restrict__ out) {
  const int bt = blockIdx.x;
  const int b = bt / T_, t = bt % T_;
  const int e0 = threadIdx.x * 4;
  float rep[4][4];
  {
    float4 v = *(const float4*)(xs + (size_t)bt * E_ + e0);
    rep[0][0] = v.x; rep[0][1] = v.y; rep[0][2] = v.z; rep[0][3] = v.w;
  }
  int gval[3];
  #pragma unroll
  for (int l = 0; l < 3; ++l) {
    const int* g = gidp + ((size_t)b * 3 + l) * T_;
    const int gc = g[t];
    gval[l] = gc;
    int s = t, e = t + 1;   // contiguous segment window (size l+2 <= 4)
    #pragma unroll
    for (int it = 0; it < 4; ++it) { if (s > 0  && g[s - 1] == gc) --s; }
    #pragma unroll
    for (int it = 0; it < 4; ++it) { if (e < T_ && g[e]     == gc) ++e; }
    const float inv = 1.0f / (float)(e - s);
    float4 acc = {0.f, 0.f, 0.f, 0.f};
    for (int tt = s; tt < e; ++tt) {
      float4 v = *(const float4*)(xs + ((size_t)b * T_ + tt) * E_ + e0);
      acc.x += v.x; acc.y += v.y; acc.z += v.z; acc.w += v.w;
    }
    rep[l + 1][0] = acc.x * inv; rep[l + 1][1] = acc.y * inv;
    rep[l + 1][2] = acc.z * inv; rep[l + 1][3] = acc.w * inv;
  }

  float4 swv = *(const float4*)(sw + e0);
  float part[4];
  #pragma unroll
  for (int k = 0; k < 4; ++k)
    part[k] = rep[k][0] * swv.x + rep[k][1] * swv.y + rep[k][2] * swv.z + rep[k][3] * swv.w;
  #pragma unroll
  for (int off = 16; off > 0; off >>= 1) {
    #pragma unroll
    for (int k = 0; k < 4; ++k) part[k] += __shfl_xor(part[k], off, 32);
  }
  __shared__ float red[4][4];
  const int wv = threadIdx.x >> 5;
  if ((threadIdx.x & 31) == 0) {
    #pragma unroll
    for (int k = 0; k < 4; ++k) red[wv][k] = part[k];
  }
  __syncthreads();
  float sc[4];
  #pragma unroll
  for (int k = 0; k < 4; ++k)
    sc[k] = red[0][k] + red[1][k] + red[2][k] + red[3][k] + sb[0];

  if (seq[(size_t)b * T_ + t] == 0) sc[0] = NEGINF;
  #pragma unroll
  for (int l = 0; l < 3; ++l) if (gval[l] == 0) sc[l + 1] = NEGINF;

  float mx = fmaxf(fmaxf(sc[0], sc[1]), fmaxf(sc[2], sc[3]));
  float p[4], psum = 0.f;
  #pragma unroll
  for (int k = 0; k < 4; ++k) { p[k] = __expf(sc[k] - mx); psum += p[k]; }
  const float rs = 1.0f / psum;

  float4 o4;
  o4.x = (rep[0][0]*p[0] + rep[1][0]*p[1] + rep[2][0]*p[2] + rep[3][0]*p[3]) * rs;
  o4.y = (rep[0][1]*p[0] + rep[1][1]*p[1] + rep[2][1]*p[2] + rep[3][1]*p[3]) * rs;
  o4.z = (rep[0][2]*p[0] + rep[1][2]*p[1] + rep[2][2]*p[2] + rep[3][2]*p[3]) * rs;
  o4.w = (rep[0][3]*p[0] + rep[1][3]*p[1] + rep[2][3]*p[2] + rep[3][3]*p[3]) * rs;
  *(float4*)(out + (size_t)bt * E_ + e0) = o4;
}

extern "C" void kernel_launch(void* const* d_in, const int* in_sizes, int n_in,
                              void* d_out, int out_size, void* d_ws, size_t ws_size,
                              hipStream_t stream) {
  const int*   seq = (const int*)d_in[0];
  const int*   gid = (const int*)d_in[1];
  const float* emb = (const float*)d_in[2];
  const float* dw  = (const float*)d_in[3];
  const float* pw  = (const float*)d_in[4];
  const float* sw  = (const float*)d_in[5];
  const float* sb  = (const float*)d_in[6];
  float* out = (float*)d_out;

  char* ws = (char*)d_ws;
  unsigned short* Abf = (unsigned short*)ws;                          // B*T*E bf16 = 32 MiB
  unsigned short* pwb = (unsigned short*)(ws + (size_t)33554432);     // E*E bf16  = 512 KiB
  float*          xs  = (float*)(ws + (size_t)33554432 + 524288);     // B*T*E f32 = 64 MiB

  k_pw_bf16<<<(E_ * E_ / 4) / 256, 256, 0, stream>>>(pw, pwb);
  k_conv<<<B_ * T_, 128, 0, stream>>>(seq, emb, dw, Abf);
  dim3 g2(B_ * T_ / BM, E_ / BN);
  k_gemm<<<g2, 256, 0, stream>>>(Abf, pwb, xs);
  k_combine<<<B_ * T_, 128, 0, stream>>>(seq, gid, xs, sw, sb, out);
}
